// FNO_5239860101553
// MI455X (gfx1250) — matrix operand order repaired and checked by reading, not compiled, not run on hardware
//
#include <hip/hip_runtime.h>
#include <hip/hip_bf16.h>
#include <math.h>

typedef _Float16 f16;
typedef _Float16 __attribute__((ext_vector_type(16))) v16h;
typedef __fp16   __attribute__((ext_vector_type(2)))  fp16x2;
typedef float    __attribute__((ext_vector_type(8)))  v8f;

#define INDIM  10
#define DIM    32
#define LAYERS 4
#define MODES  17
#define NOPS   11   // stem0,stem1, L0t,L0b,L1t,L1b,L2t,L2b,L3t,L3b, head

#define AW_HALFS  (NOPS * 32 * 16)  // 5632 f16  : A operands, per-lane layout
#define CB_FLOATS (NOPS * 32 * 8)   // 2816 f32  : bias-as-C operands, per-lane layout

// ---------------------------------------------------------------------------
// Fold kernel: collapse rfft -> mode-mix -> irfft into a real 32x32 matrix per
// layer, add proj_W, transpose, quantize to f16 in the exact WMMA A-operand
// per-lane layout. Also emit per-lane bias vectors in WMMA C/D layout.
// A[n][m] = (1/32) * sum_k c_k sum_j ( wr[j][k] cos(th*(k*m - j*n))
//                                    - wi[j][k] sin(th*(k*m - j*n)) )
// with c_0 = c_16 = 1, else 2 (irfft drops imag of DC/Nyquist), th = 2pi/32.
// ---------------------------------------------------------------------------
__global__ void fno_fold(const float* __restrict__ stem_W, const float* __restrict__ stem_b,
                         const float* __restrict__ fno_wr, const float* __restrict__ fno_wi,
                         const float* __restrict__ proj_W, const float* __restrict__ proj_b,
                         const float* __restrict__ head_W, const float* __restrict__ head_b,
                         f16* __restrict__ aw, float* __restrict__ cb)
{
  int t = blockIdx.x * blockDim.x + threadIdx.x;
  if (t < AW_HALFS) {
    int op   = t >> 9;          // / (32*16)
    int lane = (t >> 4) & 31;
    int i    = t & 15;
    int mrow = lane & 15;
    int lh   = lane >> 4;
    // 16-bit A-operand K index for (lane, half-slot i):
    int kk = lh * 16 + i;
    float val = 0.f;
    if (op < 2) {                       // stem: A[M=o][K=kk] = stem_W[kk][o], K padded
      int o = op * 16 + mrow;
      val = (kk < INDIM) ? stem_W[kk * DIM + o] : 0.f;
    } else if (op < 10) {               // layer l: A[M=o][K=n] = A_l[n][o] + proj_W[l][n][o]
      int l = (op - 2) >> 1;
      int o = ((op - 2) & 1) * 16 + mrow;
      float acc = 0.f;
      for (int k = 0; k < MODES; ++k) {
        float ck = (k == 0 || k == MODES - 1) ? 1.f : 2.f;
        for (int j = 0; j < MODES; ++j) {
          float ang = 0.19634954084936207f * (float)(k * o - j * kk);
          float s, c;
          __sincosf(ang, &s, &c);
          acc += ck * (fno_wr[(l * MODES + j) * MODES + k] * c
                     - fno_wi[(l * MODES + j) * MODES + k] * s);
        }
      }
      val = acc * (1.f / 32.f) + proj_W[(l * DIM + kk) * DIM + o];
    } else {                            // head: A[M=o][K=kk] = head_W[kk][o], M padded
      int o = mrow;
      val = (o < INDIM) ? head_W[kk * INDIM + o] : 0.f;
    }
    aw[t] = (f16)val;
  }
  int u = t - AW_HALFS;
  if (u >= 0 && u < CB_FLOATS) {        // bias in C/D layout: lane gives N, vgpr r gives M
    int op   = u >> 8;                  // / (32*8)
    int lane = (u >> 3) & 31;
    int r    = u & 7;
    int base = (lane & 16) ? 8 : 0;
    float bv = 0.f;
    if      (op == 0) bv = stem_b[base + r];
    else if (op == 1) bv = stem_b[16 + base + r];
    else if (op < 10) {
      int l = (op - 2) >> 1; int hb = (op - 2) & 1;
      bv = proj_b[l * DIM + hb * 16 + base + r];
    } else {
      int idx = base + r;
      bv = (idx < INDIM) ? head_b[idx] : 0.f;
    }
    cb[u] = bv;
  }
}

// ---------------------------------------------------------------------------
// Main fused kernel
// ---------------------------------------------------------------------------
union BU { v16h h; int i[8]; };

__device__ __forceinline__ int pkh(float a, float b) {
  fp16x2 p = __builtin_amdgcn_cvt_pkrtz(a, b);  // v_cvt_pk_rtz_f16_f32
  return __builtin_bit_cast(int, p);
}

// Convert two 16x16 f32 D tiles (features 0-15 / 16-31, C/D layout) into the
// f16 B-operand layout for the next GEMM. Select-then-swizzle: one SWAPX16
// per register pair instead of two.
__device__ __forceinline__ v16h pack_swap(v8f d0, v8f d1, bool lowhalf) {
  BU b;
#pragma unroll
  for (int j = 0; j < 4; ++j) {
    int p0 = pkh(d0[2 * j], d0[2 * j + 1]);
    int p1 = pkh(d1[2 * j], d1[2 * j + 1]);
    int t  = lowhalf ? p1 : p0;
    int u  = __builtin_amdgcn_ds_swizzle(t, 0x401F);  // SWAPX16 (xor 0x10)
    b.i[j]     = lowhalf ? p0 : u;   // low lanes: feat 0-7   | high lanes: feat 16-23
    b.i[4 + j] = lowhalf ? u  : p1;  // low lanes: feat 8-15  | high lanes: feat 24-31
  }
  return b.h;
}

// gfx1250 hardware tanh (CDNA5 trans op) -- one transcendental per SiLU.
__device__ __forceinline__ float tanh_hw(float a) {
#if __has_builtin(__builtin_amdgcn_tanhf)
  return __builtin_amdgcn_tanhf(a);
#else
  float r;
  asm("v_tanh_f32 %0, %1" : "=v"(r) : "v"(a));
  return r;
#endif
}

__device__ __forceinline__ v8f silu8(v8f v) {
#pragma unroll
  for (int r = 0; r < 8; ++r) {
    float xx = v[r];
    float t  = tanh_hw(0.5f * xx);          // sigmoid(x) = 0.5*(1+tanh(x/2))
    v[r] = xx * (0.5f * t + 0.5f);          // mul, tanh, fmaak, mul
  }
  return v;
}

__global__ void __launch_bounds__(256)
fno_main(const float* __restrict__ x, float* __restrict__ out,
         const f16* __restrict__ aw, const float* __restrict__ cbw, int ntasks)
{
  const int  lane    = threadIdx.x & 31;
  const int  gwid    = (blockIdx.x * blockDim.x + threadIdx.x) >> 5;
  const int  nw      = (gridDim.x * blockDim.x) >> 5;
  const bool lowhalf = (lane & 16) == 0;

  // One-time preload of folded weights (WMMA A operands) and biases (C operands)
  v16h A[NOPS];
  v8f  C[NOPS];
#pragma unroll
  for (int op = 0; op < NOPS; ++op) {
    const int4* pa = (const int4*)(aw + (op * 32 + lane) * 16);
    BU b;
    ((int4*)b.i)[0] = pa[0];
    ((int4*)b.i)[1] = pa[1];
    A[op] = b.h;
    const float4* pc = (const float4*)(cbw + (op * 32 + lane) * 8);
    float4 c0 = pc[0], c1 = pc[1];
    v8f cc; cc[0]=c0.x; cc[1]=c0.y; cc[2]=c0.z; cc[3]=c0.w;
            cc[4]=c1.x; cc[5]=c1.y; cc[6]=c1.z; cc[7]=c1.w;
    C[op] = cc;
  }

  for (int task = gwid; task < ntasks; task += nw) {
    const int rbase = task * 32;           // 32 rows per task = 2 x 16-row subtiles

    // ---- load 32 input rows, full wave, branchless: lane n -> row rbase+n ----
    const float2* xr = (const float2*)(x + (size_t)(rbase + lane) * INDIM);
    float2 v0 = xr[0], v1 = xr[1], v2 = xr[2], v3 = xr[3], v4 = xr[4];
    int pk[5];
    pk[0] = pkh(v0.x, v0.y); pk[1] = pkh(v1.x, v1.y); pk[2] = pkh(v2.x, v2.y);
    pk[3] = pkh(v3.x, v3.y); pk[4] = pkh(v4.x, v4.y);

    BU bx[2];
#pragma unroll
    for (int j = 0; j < 8; ++j) { bx[0].i[j] = 0; bx[1].i[j] = 0; }
#pragma unroll
    for (int j = 0; j < 5; ++j) {
      int sw = __builtin_amdgcn_ds_swizzle(pk[j], 0x401F);  // bring rows 16-31 to lanes 0-15
      bx[0].i[j] = lowhalf ? pk[j] : 0;   // subtile 0: rows 0-15 in lanes 0-15
      bx[1].i[j] = lowhalf ? sw    : 0;   // subtile 1: rows 16-31 in lanes 0-15
    }

    v16h Bh[2];

    // ---- stem: h = x @ stem_W + b (K padded 10->32) ----
#pragma unroll
    for (int s = 0; s < 2; ++s) {
      v8f d0 = __builtin_amdgcn_wmma_f32_16x16x32_f16(false, A[0], false, bx[s].h, (short)0, C[0], false, false);
      v8f d1 = __builtin_amdgcn_wmma_f32_16x16x32_f16(false, A[1], false, bx[s].h, (short)0, C[1], false, false);
      Bh[s] = pack_swap(d0, d1, lowhalf);  // no activation after stem
    }

    // ---- 4 folded (spectral + proj) layers with SiLU ----
#pragma unroll
    for (int l = 0; l < LAYERS; ++l) {
#pragma unroll
      for (int s = 0; s < 2; ++s) {
        v8f e0 = __builtin_amdgcn_wmma_f32_16x16x32_f16(false, A[2 + 2 * l], false, Bh[s], (short)0, C[2 + 2 * l], false, false);
        v8f e1 = __builtin_amdgcn_wmma_f32_16x16x32_f16(false, A[3 + 2 * l], false, Bh[s], (short)0, C[3 + 2 * l], false, false);
        e0 = silu8(e0);
        e1 = silu8(e1);
        Bh[s] = pack_swap(e0, e1, lowhalf);
      }
    }

    // ---- head: out = h @ head_W + b (N padded 10->16) ----
#pragma unroll
    for (int s = 0; s < 2; ++s) {
      v8f o = __builtin_amdgcn_wmma_f32_16x16x32_f16(false, A[10], false, Bh[s], (short)0, C[10], false, false);
      const int row = rbase + s * 16 + (lane & 15);
      float* op_ = out + (size_t)row * INDIM;
      if (lowhalf) {                       // features 0..7 of this row
        ((float2*)op_)[0] = make_float2(o[0], o[1]);
        ((float2*)op_)[1] = make_float2(o[2], o[3]);
        ((float2*)op_)[2] = make_float2(o[4], o[5]);
        ((float2*)op_)[3] = make_float2(o[6], o[7]);
      } else {                             // features 8..9
        ((float2*)(op_ + 8))[0] = make_float2(o[0], o[1]);
      }
    }

    // Prefetch next tile's inputs into cache (global_prefetch_b8)
    if (task + nw < ntasks) {
      __builtin_prefetch(x + (size_t)(task + nw) * 32 * INDIM + (size_t)lane * INDIM, 0, 1);
    }
  }
}

// ---------------------------------------------------------------------------
extern "C" void kernel_launch(void* const* d_in, const int* in_sizes, int n_in,
                              void* d_out, int out_size, void* d_ws, size_t ws_size,
                              hipStream_t stream) {
  const float* x      = (const float*)d_in[0];
  const float* stem_W = (const float*)d_in[1];
  const float* stem_b = (const float*)d_in[2];
  const float* fno_wr = (const float*)d_in[3];
  const float* fno_wi = (const float*)d_in[4];
  const float* proj_W = (const float*)d_in[5];
  const float* proj_b = (const float*)d_in[6];
  const float* head_W = (const float*)d_in[7];
  const float* head_b = (const float*)d_in[8];
  float* out = (float*)d_out;

  f16*   aw = (f16*)d_ws;
  float* cb = (float*)((char*)d_ws + (size_t)AW_HALFS * sizeof(f16));

  const int foldThreads = AW_HALFS + CB_FLOATS;   // 8448 = 33 * 256
  fno_fold<<<(foldThreads + 255) / 256, 256, 0, stream>>>(
      stem_W, stem_b, fno_wr, fno_wi, proj_W, proj_b, head_W, head_b, aw, cb);

  const int rows   = in_sizes[0] / INDIM;  // 1048576
  const int ntasks = rows / 32;            // 32768 tasks of 32 rows
  fno_main<<<512, 256, 0, stream>>>(x, out, aw, cb, ntasks);
}
